// Transition_68728066671018
// MI455X (gfx1250) — matrix operand order repaired and checked by reading, not compiled, not run on hardware
//
#include <hip/hip_runtime.h>
#include <stdint.h>

// ---------------------------------------------------------------------------
// RSSM (Dreamer-style recurrent state-space model) for MI455X / gfx1250.
// All GEMMs run on v_wmma_f32_16x16x32_bf16 (wave32 WMMA, f32 accumulate).
// Weights are converted fp32->bf16 AND pre-swizzled into WMMA-fragment-major
// layout once per launch; ~21.6MB of bf16 weights stays L2-resident (192MB L2)
// across the 50-step scan, so each lane's B fragment is two coalesced b128
// loads from contiguous memory instead of 16 strided d16 gathers.
// ---------------------------------------------------------------------------

typedef __attribute__((ext_vector_type(16))) __bf16 v16bf;
typedef __attribute__((ext_vector_type(8)))  float  v8f;

#define DEVFN static __device__ __forceinline__

DEVFN uint16_t f2bf(float f) {            // round-to-nearest-even f32 -> bf16
  union { float f; uint32_t u; } v; v.f = f;
  uint32_t r = v.u + 0x7FFFu + ((v.u >> 16) & 1u);
  return (uint16_t)(r >> 16);
}

DEVFN float sigm(float x) { return 1.0f / (1.0f + expf(-x)); }
DEVFN float softplusf(float x) { return (x > 20.0f) ? x : log1pf(expf(x)); }

DEVFN uint32_t pcg(uint32_t x) {
  x = x * 747796405u + 2891336453u;
  uint32_t w = ((x >> ((x >> 28) + 4u)) ^ x) * 277803737u;
  return (w >> 22) ^ w;
}
// deterministic standard normal via hash + Box-Muller (stand-in for jax key 42)
DEVFN float gauss(uint32_t idx, uint32_t salt) {
  uint32_t a = pcg(idx ^ (salt * 0x9E3779B9u));
  uint32_t b = pcg(a + 0x85EBCA6Bu);
  float u1 = ((a >> 8) + 1u) * (1.0f / 16777217.0f);  // (0,1]
  float u2 = (b >> 8) * (1.0f / 16777216.0f);
  return sqrtf(-2.0f * logf(u1)) * cosf(6.28318530718f * u2);
}

// ---------------------------------------------------------------------------
// Weight fp32 -> bf16 conversion + swizzle into WMMA B-fragment-major layout.
// Tile = (K=32) x (N=16). Tile order: ntile-major, ktile-minor, so a wave's
// K-loop walks contiguous 1KB tiles. Within a tile: 32 lanes x 16 bf16,
// u[j] of lane l  <->  B[ktile*32 + 16*(l>>4) + j][ntile*16 + (l&15)]
// (contiguous k order; the A fragments use the same order).
// ---------------------------------------------------------------------------
__global__ void k_swizzle_w(const float* __restrict__ W,   // K x N row-major
                            uint16_t* __restrict__ out,    // swizzled
                            int K, int N) {
  int i = blockIdx.x * blockDim.x + threadIdx.x;           // output position
  if (i >= K * N) return;
  const int j      = i & 15;
  const int lane   = (i >> 4) & 31;
  const int tile   = i >> 9;                               // /512 elems
  const int ktiles = K >> 5;
  const int ntile  = tile / ktiles;
  const int ktile  = tile - ntile * ktiles;
  const int k = ktile * 32 + 16 * (lane >> 4) + j;
  const int n = ntile * 16 + (lane & 15);
  out[i] = f2bf(W[(size_t)k * N + n]);
}

// ---------------------------------------------------------------------------
// Generic bf16 WMMA GEMM:  C[M,N] = act( A[M,K] @ Bsw[K,N] + bias[N] )
//   block = 128 threads = 4 waves; block tile = 32(M) x 64(N)
//   wave tile = 32(M) x 16(N) -> two 16x16 accumulators, 2 WMMAs per K-step
//   A tile (32x32 bf16) staged via LDS (reg double-buffered), shared by waves
//   B is pre-swizzled: per K-step each lane loads 32B contiguous (2x b128)
// Requires: M%32==0, N%64==0, K%32==0 (true for all shapes here).
// ---------------------------------------------------------------------------
template <bool RELU, bool OUTF, bool OUTB>
__global__ void __launch_bounds__(128, 2)
wmma_gemm_bf16(const uint16_t* __restrict__ A,
               const uint16_t* __restrict__ Bsw,
               const float*    __restrict__ bias,
               float*          __restrict__ Cf,   // f32 out (if OUTF)
               uint16_t*       __restrict__ Cb,   // bf16 out (if OUTB)
               int M, int N, int K)
{
  __shared__ __align__(16) uint16_t ldsA[32 * 32];

  const int lane   = threadIdx.x & 31;
  const int wv     = threadIdx.x >> 5;
  const int ntile  = blockIdx.x * 4 + wv;
  const int nBase  = ntile * 16;
  const int mBase  = blockIdx.y * 32;
  const int nn     = lane & 15;   // N (and A-row) index within tile
  const int hl     = lane >> 4;   // lane half selects K group
  const int ktiles = K >> 5;

  // this wave's swizzled-B stream: 512 elems (1KB) per K-step, contiguous
  const uint16_t* bptr = Bsw + ((size_t)ntile * ktiles * 32 + lane) * 16;

  // cooperative A tile load pattern: 128 threads x 16B covers 32x32 bf16
  const int ar = (threadIdx.x * 8) >> 5;
  const int ac = (threadIdx.x * 8) & 31;
  const uint16_t* aptr = A + (size_t)(mBase + ar) * K + ac;

  uint4 areg = *(const uint4*)aptr;   // prime the A double-buffer

  v8f acc0 = {};  // rows mBase + [0..15]
  v8f acc1 = {};  // rows mBase + [16..31]

  for (int k0 = 0; k0 < K; k0 += 32) {
    *(uint4*)(&ldsA[threadIdx.x * 8]) = areg;
    __syncthreads();

    if (k0 + 32 < K)                               // prefetch next A tile
      areg = *(const uint4*)(aptr + k0 + 32);

    // B fragment: 32 contiguous bytes per lane (wave reads 1KB contiguous)
    union { v16bf v; uint4 q[2]; uint16_t u[16]; } bf;
    bf.q[0] = *(const uint4*)(bptr);
    bf.q[1] = *(const uint4*)(bptr + 8);
    __builtin_prefetch(bptr + 512, 0, 1);          // global_prefetch next tile
    bptr += 512;

    // A fragments from LDS: lane half 0 -> K {0..15}, half 1 -> K {16..31}
    union { v16bf v; uint16_t u[16]; } af0, af1;
#pragma unroll
    for (int q = 0; q < 8; ++q) {
      const int kp = 2*q + 16 * hl;
      af0.u[2*q]     = ldsA[nn * 32 + kp];
      af0.u[2*q + 1] = ldsA[nn * 32 + kp + 1];
      af1.u[2*q]     = ldsA[(nn + 16) * 32 + kp];
      af1.u[2*q + 1] = ldsA[(nn + 16) * 32 + kp + 1];
    }

    acc0 = __builtin_amdgcn_wmma_f32_16x16x32_bf16(false, af0.v, false, bf.v,
                                                   (short)0, acc0, false, false);
    acc1 = __builtin_amdgcn_wmma_f32_16x16x32_bf16(false, af1.v, false, bf.v,
                                                   (short)0, acc1, false, false);
    __syncthreads();
  }

  const float bb = bias[nBase + nn];
#pragma unroll
  for (int q = 0; q < 8; ++q) {
    const int m0 = mBase + q + 8 * hl;      // C/D layout: row = v + 8*half
    const int m1 = m0 + 16;
    float x0 = acc0[q] + bb;
    float x1 = acc1[q] + bb;
    if (RELU) { x0 = fmaxf(x0, 0.0f); x1 = fmaxf(x1, 0.0f); }
    const size_t o0 = (size_t)m0 * N + nBase + nn;
    const size_t o1 = (size_t)m1 * N + nBase + nn;
    if (OUTF) { Cf[o0] = x0;       Cf[o1] = x1;       }
    if (OUTB) { Cb[o0] = f2bf(x0); Cb[o1] = f2bf(x1); }
  }
}

// ---------------------------------------------------------------------------
// Elementwise helpers
// ---------------------------------------------------------------------------
// carry init: post_state carry, belief carry (f32 + bf16)
__global__ void k_init_carry(const float* __restrict__ prev_state,   // B*S
                             const float* __restrict__ prev_belief,  // B*H
                             float* __restrict__ post_c,
                             float* __restrict__ belief_c,
                             uint16_t* __restrict__ belief_bf,
                             int nS, int nH) {
  int i = blockIdx.x * blockDim.x + threadIdx.x;
  if (i < nS) post_c[i] = prev_state[i];
  if (i < nH) { belief_c[i] = prev_belief[i]; belief_bf[i] = f2bf(prev_belief[i]); }
}

// x = bf16 concat([post_state * nt_t, a_t])  -> [B, S+A] = [256, 320]
__global__ void k_build_x(const float* __restrict__ post_c,   // B*S
                          const float* __restrict__ nt_t,     // B
                          const float* __restrict__ act_t,    // B*A
                          uint16_t* __restrict__ x, int n) {
  int i = blockIdx.x * blockDim.x + threadIdx.x;
  if (i >= n) return;
  const int b = i / 320;
  const int c = i - b * 320;
  float v = (c < 256) ? post_c[b * 256 + c] * nt_t[b] : act_t[b * 64 + (c - 256)];
  x[i] = f2bf(v);
}

// GRU gates (keras v2, reset_after): new belief + bf16 copies + posterior concat
__global__ void k_gru_gate(const float* __restrict__ xg,       // B x 3H
                           const float* __restrict__ hg,       // B x 3H
                           float* __restrict__ belief_c,       // B x H (in/out)
                           float* __restrict__ out_belief_t,   // B x H
                           uint16_t* __restrict__ belief_bf,   // B x H
                           uint16_t* __restrict__ concat_q,    // B x (H+O)
                           const float* __restrict__ obs_t,    // B x O
                           int n) {
  int i = blockIdx.x * blockDim.x + threadIdx.x;
  if (i >= n) return;
  const int b = i >> 10;
  const int h = i & 1023;
  const float* xr = xg + (size_t)b * 3072;
  const float* hr = hg + (size_t)b * 3072;
  const float z    = sigm(xr[h] + hr[h]);
  const float r    = sigm(xr[1024 + h] + hr[1024 + h]);
  const float cand = tanhf(xr[2048 + h] + r * hr[2048 + h]);
  const float bel  = z * belief_c[i] + (1.0f - z) * cand;
  belief_c[i]     = bel;
  out_belief_t[i] = bel;
  const uint16_t bb = f2bf(bel);
  belief_bf[i] = bb;
  concat_q[(size_t)b * 2048 + h]        = bb;
  concat_q[(size_t)b * 2048 + 1024 + h] = f2bf(obs_t[i]);
}

// head epilogue: split mean/std, softplus+MIN_STD, reparam sample
__global__ void k_head(const float* __restrict__ hd,      // B x 2S
                       float* __restrict__ out_mean_t,    // B x S
                       float* __restrict__ out_std_t,     // B x S
                       float* __restrict__ out_state_t,   // B x S
                       float* __restrict__ carry_or_null, // B x S (posterior)
                       uint32_t salt, int n) {
  int i = blockIdx.x * blockDim.x + threadIdx.x;
  if (i >= n) return;
  const int b = i >> 8;
  const int s = i & 255;
  const float m  = hd[(size_t)b * 512 + s];
  const float sd = softplusf(hd[(size_t)b * 512 + 256 + s]) + 0.1f;
  const float st = m + sd * gauss((uint32_t)i, salt);
  out_mean_t[i]  = m;
  out_std_t[i]   = sd;
  out_state_t[i] = st;
  if (carry_or_null) carry_or_null[i] = st;
}

// ---------------------------------------------------------------------------
// Host orchestration
// ---------------------------------------------------------------------------
static inline dim3 ew_grid(int n) { return dim3((n + 255) / 256); }

extern "C" void kernel_launch(void* const* d_in, const int* in_sizes, int n_in,
                              void* d_out, int out_size, void* d_ws, size_t ws_size,
                              hipStream_t stream) {
  (void)in_sizes; (void)n_in; (void)out_size; (void)ws_size;
  constexpr int T = 50, B = 256, A = 64, S = 256, H = 1024, O = 1024;
  constexpr int BH = B * H, BS = B * S;

  const float* actions   = (const float*)d_in[0];   // T,B,A
  const float* prevState = (const float*)d_in[1];   // B,S
  const float* prevBelief= (const float*)d_in[2];   // B,H
  const float* obs       = (const float*)d_in[3];   // T,B,O
  const float* nonterm   = (const float*)d_in[4];   // T,B,1
  const float* W_sa = (const float*)d_in[5];  const float* b_sa = (const float*)d_in[6];
  const float* W_x  = (const float*)d_in[7];  const float* W_h  = (const float*)d_in[8];
  const float* b_x  = (const float*)d_in[9];  const float* b_h  = (const float*)d_in[10];
  const float* W_bp = (const float*)d_in[11]; const float* b_bp = (const float*)d_in[12];
  const float* W_sp = (const float*)d_in[13]; const float* b_sp = (const float*)d_in[14];
  const float* W_bq = (const float*)d_in[15]; const float* b_bq = (const float*)d_in[16];
  const float* W_sq = (const float*)d_in[17]; const float* b_sq = (const float*)d_in[18];

  // output tuple, flat in return order
  float* out        = (float*)d_out;
  float* out_belief = out;
  float* out_prior  = out_belief + (size_t)T * BH;
  float* out_pm     = out_prior  + (size_t)T * BS;
  float* out_ps     = out_pm     + (size_t)T * BS;
  float* out_post   = out_ps     + (size_t)T * BS;
  float* out_qm     = out_post   + (size_t)T * BS;
  float* out_qs     = out_qm     + (size_t)T * BS;

  // workspace bump allocator (256B aligned)
  char* wp = (char*)d_ws;
  auto alloc = [&](size_t bytes) -> void* {
    void* p = (void*)wp;
    wp += (bytes + 255) & ~(size_t)255;
    return p;
  };
  uint16_t* Wsa_bf = (uint16_t*)alloc(sizeof(uint16_t) * 320 * 1024);
  uint16_t* Wx_bf  = (uint16_t*)alloc(sizeof(uint16_t) * 1024 * 3072);
  uint16_t* Wh_bf  = (uint16_t*)alloc(sizeof(uint16_t) * 1024 * 3072);
  uint16_t* Wbp_bf = (uint16_t*)alloc(sizeof(uint16_t) * 1024 * 1024);
  uint16_t* Wsp_bf = (uint16_t*)alloc(sizeof(uint16_t) * 1024 * 512);
  uint16_t* Wbq_bf = (uint16_t*)alloc(sizeof(uint16_t) * 2048 * 1024);
  uint16_t* Wsq_bf = (uint16_t*)alloc(sizeof(uint16_t) * 1024 * 512);
  uint16_t* x_bf     = (uint16_t*)alloc(sizeof(uint16_t) * B * 320);
  uint16_t* h1_bf    = (uint16_t*)alloc(sizeof(uint16_t) * BH);
  uint16_t* belief_bf= (uint16_t*)alloc(sizeof(uint16_t) * BH);
  uint16_t* concat_bf= (uint16_t*)alloc(sizeof(uint16_t) * B * 2048);
  uint16_t* p1_bf    = (uint16_t*)alloc(sizeof(uint16_t) * BH);
  uint16_t* q1_bf    = (uint16_t*)alloc(sizeof(uint16_t) * BH);
  float* xg_f     = (float*)alloc(sizeof(float) * B * 3072);
  float* hg_f     = (float*)alloc(sizeof(float) * B * 3072);
  float* belief_c = (float*)alloc(sizeof(float) * BH);
  float* post_c   = (float*)alloc(sizeof(float) * BS);
  float* ph_f     = (float*)alloc(sizeof(float) * B * 512);
  float* qh_f     = (float*)alloc(sizeof(float) * B * 512);

  // one-time weight conversion + fragment swizzle (then L2-resident all scan)
  k_swizzle_w<<<ew_grid(320 * 1024),  256, 0, stream>>>(W_sa, Wsa_bf, 320, 1024);
  k_swizzle_w<<<ew_grid(1024 * 3072), 256, 0, stream>>>(W_x,  Wx_bf,  1024, 3072);
  k_swizzle_w<<<ew_grid(1024 * 3072), 256, 0, stream>>>(W_h,  Wh_bf,  1024, 3072);
  k_swizzle_w<<<ew_grid(1024 * 1024), 256, 0, stream>>>(W_bp, Wbp_bf, 1024, 1024);
  k_swizzle_w<<<ew_grid(1024 * 512),  256, 0, stream>>>(W_sp, Wsp_bf, 1024, 512);
  k_swizzle_w<<<ew_grid(2048 * 1024), 256, 0, stream>>>(W_bq, Wbq_bf, 2048, 1024);
  k_swizzle_w<<<ew_grid(1024 * 512),  256, 0, stream>>>(W_sq, Wsq_bf, 1024, 512);
  k_init_carry<<<ew_grid(BH), 256, 0, stream>>>(prevState, prevBelief,
                                                post_c, belief_c, belief_bf, BS, BH);

  for (int t = 0; t < T; ++t) {
    const float* act_t = actions + (size_t)t * B * A;
    const float* obs_t = obs     + (size_t)t * B * O;
    const float* nt_t  = nonterm + (size_t)t * B;

    // x = [post*nt, a]  (B x 320 bf16)
    k_build_x<<<ew_grid(B * 320), 256, 0, stream>>>(post_c, nt_t, act_t, x_bf, B * 320);
    // h1 = relu(x @ W_sa + b_sa)            (256x1024, K=320) -> bf16
    wmma_gemm_bf16<true, false, true ><<<dim3(1024 / 64, 256 / 32), 128, 0, stream>>>(
        x_bf, Wsa_bf, b_sa, nullptr, h1_bf, 256, 1024, 320);
    // xg = h1 @ W_x + b_x                   (256x3072, K=1024) -> f32
    wmma_gemm_bf16<false, true, false><<<dim3(3072 / 64, 256 / 32), 128, 0, stream>>>(
        h1_bf, Wx_bf, b_x, xg_f, nullptr, 256, 3072, 1024);
    // hg = belief @ W_h + b_h               (256x3072, K=1024) -> f32
    wmma_gemm_bf16<false, true, false><<<dim3(3072 / 64, 256 / 32), 128, 0, stream>>>(
        belief_bf, Wh_bf, b_h, hg_f, nullptr, 256, 3072, 1024);
    // GRU gates -> belief (f32 carry + out + bf16) + posterior concat [belief,o]
    k_gru_gate<<<ew_grid(BH), 256, 0, stream>>>(
        xg_f, hg_f, belief_c, out_belief + (size_t)t * BH,
        belief_bf, concat_bf, obs_t, BH);
    // prior head
    wmma_gemm_bf16<true, false, true ><<<dim3(1024 / 64, 256 / 32), 128, 0, stream>>>(
        belief_bf, Wbp_bf, b_bp, nullptr, p1_bf, 256, 1024, 1024);
    wmma_gemm_bf16<false, true, false><<<dim3(512 / 64, 256 / 32), 128, 0, stream>>>(
        p1_bf, Wsp_bf, b_sp, ph_f, nullptr, 256, 512, 1024);
    k_head<<<ew_grid(BS), 256, 0, stream>>>(
        ph_f, out_pm + (size_t)t * BS, out_ps + (size_t)t * BS,
        out_prior + (size_t)t * BS, nullptr, (uint32_t)(2 * t + 0), BS);
    // posterior head (K=2048 concat GEMM)
    wmma_gemm_bf16<true, false, true ><<<dim3(1024 / 64, 256 / 32), 128, 0, stream>>>(
        concat_bf, Wbq_bf, b_bq, nullptr, q1_bf, 256, 1024, 2048);
    wmma_gemm_bf16<false, true, false><<<dim3(512 / 64, 256 / 32), 128, 0, stream>>>(
        q1_bf, Wsq_bf, b_sq, qh_f, nullptr, 256, 512, 1024);
    k_head<<<ew_grid(BS), 256, 0, stream>>>(
        qh_f, out_qm + (size_t)t * BS, out_qs + (size_t)t * BS,
        out_post + (size_t)t * BS, post_c, (uint32_t)(2 * t + 1), BS);
  }
}
